// SRUModel_17721035063985
// MI455X (gfx1250) — compile-verified, matrix operand order checked by reading, not run on hardware
//
#include <hip/hip_runtime.h>
#include <hip/hip_bf16.h>

typedef __attribute__((ext_vector_type(16))) _Float16 v16h;
typedef __attribute__((ext_vector_type(8)))  _Float16 v8h;
typedef __attribute__((ext_vector_type(4)))  _Float16 v4h;
typedef __attribute__((ext_vector_type(8)))  float    v8f;

#define VOCAB   50000
#define EMBED   300
#define HIDDEN  128
#define BATCH   64
#define SEQ     1024
#define NCOL    512      // 4*HIDDEN
#define KPAD    320      // EMBED padded to 10 chunks of 32
#define KCHUNKS 10
#define TCHUNK  16       // timesteps per WMMA/scan chunk
#define NCHUNKS (SEQ / TCHUNK)
#define QUADS   (EMBED / 4)   // 75 float4 per embedding row

// ---------------------------------------------------------------------------
// Kernel 1: transpose + convert W (300x512 f32, K-major) into Wt (512x320 f16,
// N-major, K zero-padded). 327 KB -> L2 resident for the fused kernel.
// ---------------------------------------------------------------------------
__global__ __launch_bounds__(256) void wt_prep_kernel(const float* __restrict__ W,
                                                      _Float16* __restrict__ Wt) {
    int i = blockIdx.x * 256 + threadIdx.x;          // over 512*320
    if (i >= NCOL * KPAD) return;
    int n = i / KPAD;
    int k = i - n * KPAD;
    float v = (k < EMBED) ? W[k * NCOL + n] : 0.0f;
    Wt[i] = (_Float16)v;
}

__device__ __forceinline__ float sigmoidf(float x) {
    return 1.0f / (1.0f + __expf(-x));
}

__device__ __forceinline__ v16h load_frag16(const _Float16* p) {
    // halves [0..7] and [16..23] relative to p (16-byte aligned)
    v8h lo = *(const v8h*)(p);
    v8h hi = *(const v8h*)(p + 16);
    return __builtin_shufflevector(lo, hi, 0, 1, 2, 3, 4, 5, 6, 7,
                                           8, 9, 10, 11, 12, 13, 14, 15);
}

// Gather one 16-token x 300-K A tile (f32 -> f16) into LDS; K tail pre-zeroed.
__device__ __forceinline__ void fill_a_tile(const int* __restrict__ x,
                                            const float* __restrict__ emb,
                                            _Float16* __restrict__ dstA,
                                            int b, int t0, int tid) {
    for (int i = tid; i < TCHUNK * QUADS; i += 256) {
        int row = i / QUADS;
        int q   = i - row * QUADS;
        int tok = x[b * SEQ + t0 + row];
        float4 v = *(const float4*)(emb + (size_t)tok * EMBED + q * 4);
        v4h hv = { (_Float16)v.x, (_Float16)v.y, (_Float16)v.z, (_Float16)v.w };
        *(v4h*)(dstA + row * KPAD + q * 4) = hv;    // 8B-aligned ds_store_b64
    }
}

// 16 sequential SRU steps from an LDS U chunk.
__device__ __forceinline__ void scan16(const float* __restrict__ ub,
                                       float vf, float vr, float bf, float br,
                                       int h, float& c, float& hsum, float& hmax) {
    #pragma unroll
    for (int tt = 0; tt < TCHUNK; ++tt) {
        const float* up = ub + tt * NCOL;
        float u0 = up[h];
        float u1 = up[HIDDEN + h];
        float u2 = up[2 * HIDDEN + h];
        float u3 = up[3 * HIDDEN + h];
        float f  = sigmoidf(u1 + vf * c + bf);
        float r_ = sigmoidf(u2 + vr * c + br);
        c = f * c + (1.0f - f) * u0;
        float ho = r_ * c + (1.0f - r_) * u3;
        hsum += ho;
        hmax = fmaxf(hmax, ho);
    }
}

// ---------------------------------------------------------------------------
// Kernel 2: fused gather -> WMMA GEMM -> SRU scan -> pool -> fc -> sigmoid.
// One block per batch element; 256 threads = 8 waves. Software-pipelined with
// double-buffered A and U tiles: one barrier per 16-timestep chunk. The B
// matrix (this wave's 64 Wt columns, 40 v16h fragments = 320 VGPRs) is
// loop-invariant and held entirely in registers, so the steady-state loop is
// ds_load(A) + v_wmma + scan VALU + gather VMEM with no global B traffic.
// ---------------------------------------------------------------------------
__global__ __launch_bounds__(256) void sru_fused_kernel(
    const int*      __restrict__ x,     // (64,1024)
    const float*    __restrict__ emb,   // (50000,300)
    const _Float16* __restrict__ Wt,    // (512,320) f16
    const float*    __restrict__ bvec,  // (256,)
    const float*    __restrict__ wc,    // (256,)
    const float*    __restrict__ fcw,   // (256,)
    const float*    __restrict__ fcb,   // (1,)
    float*          __restrict__ out)   // (64,)
{
    __shared__ _Float16 ldsA[2][TCHUNK * KPAD];   // 2 x 10 KB
    __shared__ float    ldsU[2][TCHUNK * NCOL];   // 2 x 32 KB
    __shared__ float    red[128];

    const int b    = blockIdx.x;
    const int tid  = threadIdx.x;
    const int lane = tid & 31;
    const int wave = tid >> 5;
    const int n0   = wave * 64;                 // this wave's 64-column slab

    // Fragment addressing (ISA 16-bit A 16x32 layout; B mirrored from Wt)
    const int lrow  = lane & 15;
    const int lkoff = (lane >> 4) * 8;

    // per-thread SRU state (threads 0..127 own hidden unit h = tid)
    const int h = tid;
    float vf = 0.f, vr = 0.f, bf = 0.f, br = 0.f;
    if (h < HIDDEN) {
        vf = wc[h];           vr = wc[HIDDEN + h];
        bf = bvec[h];         br = bvec[HIDDEN + h];
    }
    float c = 0.0f, hsum = 0.0f, hmax = -3.402823466e38f;

    // ---- hoist the loop-invariant B fragments into registers ----
    const _Float16* Bbase = &Wt[(size_t)(n0 + lrow) * KPAD + lkoff];
    v16h Bf[KCHUNKS][4];
    #pragma unroll
    for (int kc = 0; kc < KCHUNKS; ++kc) {
        #pragma unroll
        for (int t = 0; t < 4; ++t)
            Bf[kc][t] = load_frag16(Bbase + t * 16 * KPAD + kc * 32);
    }

    // zero K-pad tail (cols 300..319) of BOTH A buffers once
    if (tid < 160) {
        int buf = tid / 80, rem = tid - buf * 80;
        int row = rem / 5,  j   = rem - row * 5;
        *(uint2*)&ldsA[buf][row * KPAD + EMBED + j * 4] = make_uint2(0u, 0u);
    }
    // prologue: stage A tile for chunk 0
    fill_a_tile(x, emb, ldsA[0], b, 0, tid);
    __syncthreads();

    for (int i = 0; i < NCHUNKS; ++i) {
        const int abuf = i & 1;

        // ---- WMMA for chunk i : 16(M) x 64(N) per wave, K = 10 x 32 ----
        v8f acc0 = {}, acc1 = {}, acc2 = {}, acc3 = {};
        const _Float16* Abase = &ldsA[abuf][lrow * KPAD + lkoff];
        #pragma unroll
        for (int kc = 0; kc < KCHUNKS; ++kc) {
            v16h af = load_frag16(Abase + kc * 32);
            acc0 = __builtin_amdgcn_wmma_f32_16x16x32_f16(false, af, false, Bf[kc][0],
                                                          (short)0, acc0, false, false);
            acc1 = __builtin_amdgcn_wmma_f32_16x16x32_f16(false, af, false, Bf[kc][1],
                                                          (short)0, acc1, false, false);
            acc2 = __builtin_amdgcn_wmma_f32_16x16x32_f16(false, af, false, Bf[kc][2],
                                                          (short)0, acc2, false, false);
            acc3 = __builtin_amdgcn_wmma_f32_16x16x32_f16(false, af, false, Bf[kc][3],
                                                          (short)0, acc3, false, false);
        }

        // ---- overlap: gather next chunk's A tile (VMEM || XDL) ----
        if (i + 1 < NCHUNKS)
            fill_a_tile(x, emb, ldsA[abuf ^ 1], b, (i + 1) * TCHUNK, tid);

        // ---- overlap: scan chunk i-1 (VALU/TRANS co-executes with WMMA) ----
        if (i > 0 && h < HIDDEN)
            scan16(ldsU[(i - 1) & 1], vf, vr, bf, br, h, c, hsum, hmax);

        // ---- drain WMMA: store D tiles (M = r + 8*(lane>=16), N = lane%16) ----
        {
            const int mbase = (lane >> 4) << 3;
            const int col   = n0 + (lane & 15);
            float* ub = ldsU[abuf];
            #pragma unroll
            for (int r = 0; r < 8; ++r) {
                float* dst = &ub[(mbase + r) * NCOL + col];
                dst[ 0] = acc0[r];
                dst[16] = acc1[r];
                dst[32] = acc2[r];
                dst[48] = acc3[r];
            }
        }
        __syncthreads();   // U[i] published; A[i+1] published; old buffers free
    }

    // epilogue: scan the final chunk
    if (h < HIDDEN)
        scan16(ldsU[(NCHUNKS - 1) & 1], vf, vr, bf, br, h, c, hsum, hmax);

    // pooled = [avg, max]; logit = pooled . fc_w + fc_b; sigmoid
    if (h < HIDDEN) {
        red[h] = (hsum * (1.0f / (float)SEQ)) * fcw[h] + hmax * fcw[HIDDEN + h];
    }
    __syncthreads();
    if (tid < 32) {
        float v = red[tid] + red[tid + 32] + red[tid + 64] + red[tid + 96];
        #pragma unroll
        for (int off = 16; off >= 1; off >>= 1)
            v += __shfl_xor(v, off, 32);
        if (tid == 0)
            out[b] = sigmoidf(v + fcb[0]);
    }
}

extern "C" void kernel_launch(void* const* d_in, const int* in_sizes, int n_in,
                              void* d_out, int out_size, void* d_ws, size_t ws_size,
                              hipStream_t stream) {
    (void)in_sizes; (void)n_in; (void)out_size; (void)ws_size;
    const int*   x    = (const int*)  d_in[0];
    const float* emb  = (const float*)d_in[1];
    const float* W    = (const float*)d_in[2];
    const float* bvec = (const float*)d_in[3];
    const float* wc   = (const float*)d_in[4];
    const float* fcw  = (const float*)d_in[5];
    const float* fcb  = (const float*)d_in[6];
    float*       out  = (float*)d_out;

    _Float16* Wt = (_Float16*)d_ws;   // 512*320*2 = 327,680 bytes

    wt_prep_kernel<<<(NCOL * KPAD + 255) / 256, 256, 0, stream>>>(W, Wt);
    sru_fused_kernel<<<BATCH, 256, 0, stream>>>(x, emb, Wt, bvec, wc, fcw, fcb, out);
}
